// SSIM_46050639348111
// MI455X (gfx1250) — matrix operand order repaired and checked by reading, not compiled, not run on hardware
//
#include <hip/hip_runtime.h>

typedef __attribute__((ext_vector_type(16))) _Float16 v16h;
typedef __attribute__((ext_vector_type(8)))  _Float16 v8h;
typedef __attribute__((ext_vector_type(4)))  _Float16 v4h;
typedef __attribute__((ext_vector_type(8)))  float    v8f;
typedef __attribute__((ext_vector_type(4)))  float    v4f;

#define IMG_H 512
#define IMG_W 512
#define NPLANES 48            // 16 batch * 3 channels
#define TILES_PER_PLANE 1024  // 32 x 32 tiles of 16x16
#define WROWS 26              // 16 + 2*5 halo rows
#define LDS_HALVES_PER_WAVE (1024 + 1024 + 512)  // xTile 32x32, yTile 32x32, inter 16x32

__device__ __constant__ float c_gk[11] = {
    0.00102838f, 0.00759876f, 0.03600094f, 0.10936078f, 0.21300560f,
    0.26601260f,
    0.21300560f, 0.10936078f, 0.03600094f, 0.00759876f, 0.00102838f
};

__global__ __launch_bounds__(256) void ssim_wmma_kernel(
    const float* __restrict__ img1, const float* __restrict__ img2,
    float* __restrict__ out, float invN)
{
    __shared__ _Float16 smem[8 * LDS_HALVES_PER_WAVE];
    __shared__ float wsum[8];

    const int lane = threadIdx.x & 31;
    const int wid  = threadIdx.x >> 5;
    const int waveId = blockIdx.x * 8 + wid;
    const int plane  = waveId >> 10;          // / TILES_PER_PLANE
    const int t      = waveId & 1023;
    const int ty = t >> 5, tx = t & 31;

    _Float16* xT = smem + wid * LDS_HALVES_PER_WAVE;  // 32x32 f16, row-major
    _Float16* yT = xT + 1024;                         // 32x32 f16, row-major
    _Float16* iT = yT + 1024;                         // inter, [col][k] 16x32 f16

    const float* p1 = img1 + (size_t)plane * (IMG_H * IMG_W);
    const float* p2 = img2 + (size_t)plane * (IMG_H * IMG_W);
    const int row0 = ty * 16 - 5;   // vertical window start (taps -5..+5)
    const int col0 = tx * 16 - 8;   // horizontal window start, 32B-aligned

    const float C1f = 1.0e-4f;
    const float C2f = 9.0e-4f;

    // ---- Stage halo window into LDS (rows 0..25 valid, rows 26..31 zeroed) ----
    const bool interior = (tx >= 1) && (tx <= 30) && (ty >= 1) && (ty <= 30);
    if (interior) {
        // Fast path: 16B-aligned float4 loads, no bounds checks (wave-uniform).
        const float* r1 = p1 + (size_t)row0 * IMG_W + col0;
        const float* r2 = p2 + (size_t)row0 * IMG_W + col0;
        #pragma unroll
        for (int j = 0; j < 8; ++j) {
            int e4 = j * 32 + lane;      // float4 slot in 32(row) x 8(chunk) grid
            int r  = e4 >> 3;
            int c4 = (e4 & 7) << 2;
            v4f xv = {}, yv = {};
            if (r < WROWS) {
                xv = *(const v4f*)(r1 + r * IMG_W + c4);
                yv = *(const v4f*)(r2 + r * IMG_W + c4);
            }
            v4h xh, yh;
            #pragma unroll
            for (int k = 0; k < 4; ++k) {
                xh[k] = (_Float16)xv[k];
                yh[k] = (_Float16)yv[k];
            }
            *(v4h*)(xT + r * 32 + c4) = xh;
            *(v4h*)(yT + r * 32 + c4) = yh;
        }
    } else {
        // Edge tiles: per-element zero-padded loads.
        #pragma unroll
        for (int idx = 0; idx < 32; ++idx) {
            int e = idx * 32 + lane;
            int r = e >> 5, c = e & 31;
            int gr = row0 + r, gc = col0 + c;
            float xv = 0.0f, yv = 0.0f;
            if (r < WROWS && gr >= 0 && gr < IMG_H && gc >= 0 && gc < IMG_W) {
                size_t off = (size_t)gr * IMG_W + gc;
                xv = p1[off];
                yv = p2[off];
            }
            xT[e] = (_Float16)xv;
            yT[e] = (_Float16)yv;
        }
    }

    // ---- Per-lane constant band fragments (documented 16-bit WMMA layouts) ----
    const int Nl    = lane & 15;     // col (B/C/D) or row M (A)
    const int hi    = lane >> 4;     // lane half
    const int baseK = hi ? 8 : 0;    // A-matrix K base for this lane half

    // B_h (32x16): lane holds K = baseK + i (i<8) and baseK + 16 + (i-8), col = Nl.
    // Window starts at n-8, tap t at k = n + 3 + t  =>  B[k][n] = g[k-n-3], k-n in [3,13]
    v16h Bh;
    #pragma unroll
    for (int i = 0; i < 16; ++i) {
        int d = (baseK + i + (i >= 8 ? 8 : 0)) - Nl;
        Bh[i] = (_Float16)((d >= 3 && d <= 13) ? c_gk[d - 3] : 0.0f);
    }
    // A_v (16x32): lane row M = Nl, K = baseK+i (i<8) and baseK+16+i ; A[m][k] = g[k-m]
    v16h Av;
    #pragma unroll
    for (int i = 0; i < 8; ++i) {
        int d0 = (baseK + i) - Nl;
        int d1 = (baseK + 16 + i) - Nl;
        Av[i]     = (_Float16)((d0 >= 0 && d0 <= 10) ? c_gk[d0] : 0.0f);
        Av[8 + i] = (_Float16)((d1 >= 0 && d1 <= 10) ? c_gk[d1] : 0.0f);
    }

    // ---- Fetch per-lane A-matrix data chunks (two row-groups, two K-chunks) ----
    v8h xc[2][2], yc[2][2];
    #pragma unroll
    for (int g = 0; g < 2; ++g) {
        int row = g * 16 + Nl;
        xc[g][0] = *(const v8h*)(xT + row * 32 + baseK);
        xc[g][1] = *(const v8h*)(xT + row * 32 + baseK + 16);
        yc[g][0] = *(const v8h*)(yT + row * 32 + baseK);
        yc[g][1] = *(const v8h*)(yT + row * 32 + baseK + 16);
    }

    // ---- Five maps: x, y, x*x, y*y, x*y ; each = 2 H-WMMA + 1 V-WMMA ----
    v8f res[5];
    #pragma unroll
    for (int m = 0; m < 5; ++m) {
        #pragma unroll
        for (int g = 0; g < 2; ++g) {
            v8h a0, a1;
            if      (m == 0) { a0 = xc[g][0];            a1 = xc[g][1]; }
            else if (m == 1) { a0 = yc[g][0];            a1 = yc[g][1]; }
            else if (m == 2) { a0 = xc[g][0] * xc[g][0]; a1 = xc[g][1] * xc[g][1]; }
            else if (m == 3) { a0 = yc[g][0] * yc[g][0]; a1 = yc[g][1] * yc[g][1]; }
            else             { a0 = xc[g][0] * yc[g][0]; a1 = xc[g][1] * yc[g][1]; }
            v16h A = __builtin_shufflevector(a0, a1,
                0,1,2,3,4,5,6,7,8,9,10,11,12,13,14,15);
            v8f ch = {};
            ch = __builtin_amdgcn_wmma_f32_16x16x32_f16(
                false, A, false, Bh, (short)0, ch, false, false);
            // Store transposed into inter: iT[col][k], k = g*16 + baseM + i
            v8h hv;
            #pragma unroll
            for (int i = 0; i < 8; ++i) hv[i] = (_Float16)ch[i];
            *(v8h*)(iT + Nl * 32 + g * 16 + baseK) = hv;
        }
        // Vertical pass: B_v lane = inter[K][Nl], K = hi*8 + 0..7 and 16 + hi*8 + 0..7
        v8h b0 = *(const v8h*)(iT + Nl * 32 + hi * 8);
        v8h b1 = *(const v8h*)(iT + Nl * 32 + hi * 8 + 16);
        v16h Bv = __builtin_shufflevector(b0, b1,
            0,1,2,3,4,5,6,7,8,9,10,11,12,13,14,15);
        v8f cv = {};
        res[m] = __builtin_amdgcn_wmma_f32_16x16x32_f16(
            false, Av, false, Bv, (short)0, cv, false, false);
    }

    // ---- SSIM map + loss over the 8 elements this lane owns ----
    float acc = 0.0f;
    #pragma unroll
    for (int i = 0; i < 8; ++i) {
        float m1 = res[0][i], m2 = res[1][i];
        float m1s = m1 * m1, m2s = m2 * m2, m12 = m1 * m2;
        float s1  = res[2][i] - m1s;
        float s2  = res[3][i] - m2s;
        float s12 = res[4][i] - m12;
        float num = (2.0f * m12 + C1f) * (2.0f * s12 + C2f);
        float den = (m1s + m2s + C1f) * (s1 + s2 + C2f);
        float loss = 1.0f - num / den;
        loss = fminf(fmaxf(loss, 0.0f), 1.0f) * 0.5f;
        acc += loss;
    }

    // ---- wave32 reduction, block reduction, one atomic per block ----
    #pragma unroll
    for (int off = 16; off > 0; off >>= 1)
        acc += __shfl_down(acc, off, 32);
    if (lane == 0) wsum[wid] = acc;
    __syncthreads();
    if (wid == 0) {
        float v = (lane < 8) ? wsum[lane] : 0.0f;
        #pragma unroll
        for (int off = 4; off > 0; off >>= 1)
            v += __shfl_down(v, off, 32);
        if (lane == 0)
            atomicAdd(out, v * invN);
    }
}

extern "C" void kernel_launch(void* const* d_in, const int* in_sizes, int n_in,
                              void* d_out, int out_size, void* d_ws, size_t ws_size,
                              hipStream_t stream) {
    (void)in_sizes; (void)n_in; (void)out_size; (void)d_ws; (void)ws_size;
    const float* img1 = (const float*)d_in[0];
    const float* img2 = (const float*)d_in[1];
    float* out = (float*)d_out;

    hipMemsetAsync(out, 0, sizeof(float), stream);

    const float invN = 1.0f / (float)(16 * 3 * 512 * 512);
    const int totalWaves = NPLANES * TILES_PER_PLANE;  // 49152 tiles
    const int blocks     = totalWaves / 8;             // 6144 blocks x 8 waves
    ssim_wmma_kernel<<<blocks, 256, 0, stream>>>(img1, img2, out, invN);
}
